// _HybridQLSTMCell_65481071408609
// MI455X (gfx1250) — matrix operand order repaired and checked by reading, not compiled, not run on hardware
//
#include <hip/hip_runtime.h>

// ---------------------------------------------------------------------------
// QLSTM cell, fused single pass.
//   gate = [x|hx] @ W^T + b            (WMMA f32<-f16, K=68 in 3 tiles of 32)
//   E^t  = parity products of cos(gate_w + theta_t_w)   (closed form of the
//          RX product state + CNOT ring + <Z> measurement)
//   cx'  = sig(Ef)*cx + sig(Ei)*tanh(Eg);  hx' = sig(Eo)*tanh(cx')
// Bandwidth-bound: ~160 MB traffic -> ~7 us floor @ 23.3 TB/s.
// ---------------------------------------------------------------------------

typedef __attribute__((ext_vector_type(16))) _Float16 v16h;
typedef __attribute__((ext_vector_type(8)))  float    v8f;

#define QB   524288
#define QIN  64
#define QH   4
#define KTOT 68   // IN + H

// B fragment (32x16 f16): lane = column N; elems 0-7 = K kb..kb+7, 8-15 = K kb+16..kb+23; lanes 16-31 kb += 8.
// B[k][n] = W[n][k]; columns n>=4 and k>=68 are zero padding.
__device__ __forceinline__ v16h make_b_frag(const float* __restrict__ W, int lane, int t) {
  const int n = lane & 15;
  const int kbase = t * 32 + ((lane & 16) ? 8 : 0);
  v16h b;
#pragma unroll
  for (int e = 0; e < 16; ++e) {
    const int k = kbase + (e < 8 ? e : e + 8);
    const bool valid = (n < QH) && (k < KTOT);
    const float v = W[valid ? (n * KTOT + k) : 0];
    b[e] = valid ? (_Float16)v : (_Float16)0.0f;
  }
  return b;
}

// A fragment (16x32 f16) for K-tiles 0/1 (all K < 64 -> from x).
// lane<16: M=lane, elems 0-7 = K kb..kb+7, elems 8-15 = K kb+16..kb+23 (kb = t*32)
// lane>=16: same with kb += 8.
__device__ __forceinline__ v16h make_a_frag01(const float* __restrict__ x,
                                              int rbase, int lane, int t) {
  const int m = lane & 15;
  const float* base = x + (size_t)(rbase + m) * QIN + t * 32 + ((lane & 16) ? 8 : 0);
  const float4 f0 = ((const float4*)base)[0];
  const float4 f1 = ((const float4*)base)[1];
  const float4 f2 = ((const float4*)(base + 16))[0];
  const float4 f3 = ((const float4*)(base + 16))[1];
  v16h a;
  a[0]=(_Float16)f0.x;  a[1]=(_Float16)f0.y;  a[2]=(_Float16)f0.z;  a[3]=(_Float16)f0.w;
  a[4]=(_Float16)f1.x;  a[5]=(_Float16)f1.y;  a[6]=(_Float16)f1.z;  a[7]=(_Float16)f1.w;
  a[8]=(_Float16)f2.x;  a[9]=(_Float16)f2.y;  a[10]=(_Float16)f2.z; a[11]=(_Float16)f2.w;
  a[12]=(_Float16)f3.x; a[13]=(_Float16)f3.y; a[14]=(_Float16)f3.z; a[15]=(_Float16)f3.w;
  return a;
}

// A fragment for K-tile 2: only K=64..67 (hx) nonzero, which lands in
// elements 0..3 of lanes 0-15 (K 64..71); everything else is zero.
__device__ __forceinline__ v16h make_a_frag2(const float* __restrict__ hx,
                                             int rbase, int lane) {
  const int m = lane & 15;
  const float4 h = *(const float4*)(hx + (size_t)(rbase + m) * QH);
  const bool lo = !(lane & 16);
  v16h a;
#pragma unroll
  for (int e = 0; e < 16; ++e) a[e] = (_Float16)0.0f;
  a[0] = lo ? (_Float16)h.x : (_Float16)0.0f;
  a[1] = lo ? (_Float16)h.y : (_Float16)0.0f;
  a[2] = lo ? (_Float16)h.z : (_Float16)0.0f;
  a[3] = lo ? (_Float16)h.w : (_Float16)0.0f;
  return a;
}

// Closed-form <Z_w> after CNOT ring: parities of d_w = cos(gate_w + theta_w).
__device__ __forceinline__ float4 qexpect(float g0, float g1, float g2, float g3,
                                          const float* __restrict__ th) {
  const float d0 = __cosf(g0 + th[0]);
  const float d1 = __cosf(g1 + th[1]);
  const float d2 = __cosf(g2 + th[2]);
  const float d3 = __cosf(g3 + th[3]);
  float4 e;
  e.x = d1 * d2 * d3;   // parity {1,2,3}
  e.y = d0 * d1;        // parity {0,1}
  e.z = e.y * d2;       // parity {0,1,2}
  e.w = e.z * d3;       // parity {0,1,2,3}
  return e;
}

__device__ __forceinline__ float sig_(float v)  { return 1.0f / (1.0f + __expf(-v)); }
__device__ __forceinline__ float tanh_(float v) {
  const float e = __expf(2.0f * v);
  return (e - 1.0f) / (e + 1.0f);   // inputs bounded (|E|<=1, cx ~ N(0,1)); safe
}

__global__ __launch_bounds__(256) void qlstm_kernel(
    const float* __restrict__ x,  const float* __restrict__ hx,
    const float* __restrict__ cx, const float* __restrict__ W,
    const float* __restrict__ bias,
    const float* __restrict__ th_f, const float* __restrict__ th_i,
    const float* __restrict__ th_g, const float* __restrict__ th_o,
    float* __restrict__ out)
{
  __shared__ float sgate[8][32][QH];   // per-wave gate_vec transpose, 4 KB

  const int tid      = threadIdx.x;
  const int wave     = tid >> 5;
  const int lane     = tid & 31;
  const int blockRow = blockIdx.x * 256;
  const int waveRow  = blockRow + wave * 32;

  // --- Phase 1: gate_vec GEMM via WMMA --------------------------------------
  const v16h b0 = make_b_frag(W, lane, 0);
  const v16h b1 = make_b_frag(W, lane, 1);
  const v16h b2 = make_b_frag(W, lane, 2);

  const int   n  = lane & 15;
  const float bn = (n < QH) ? bias[n] : 0.0f;

#pragma unroll
  for (int half = 0; half < 2; ++half) {     // two 16-row tiles per wave
    const int rbase = waveRow + half * 16;
    const v16h a0 = make_a_frag01(x, rbase, lane, 0);
    const v16h a1 = make_a_frag01(x, rbase, lane, 1);
    const v16h a2 = make_a_frag2(hx, rbase, lane);

    v8f acc = {0.f, 0.f, 0.f, 0.f, 0.f, 0.f, 0.f, 0.f};
    acc = __builtin_amdgcn_wmma_f32_16x16x32_f16(false, a0, false, b0, (short)0, acc, false, false);
    acc = __builtin_amdgcn_wmma_f32_16x16x32_f16(false, a1, false, b1, (short)0, acc, false, false);
    acc = __builtin_amdgcn_wmma_f32_16x16x32_f16(false, a2, false, b2, (short)0, acc, false, false);

    // C layout: lane holds column n for rows mbase..mbase+7 -> transpose via LDS
    if (n < QH) {
      const int mbase = half * 16 + ((lane & 16) ? 8 : 0);
#pragma unroll
      for (int j = 0; j < 8; ++j)
        sgate[wave][mbase + j][n] = acc[j] + bn;
    }
  }
  __syncthreads();

  // --- Phase 2: closed-form quantum layers + LSTM tail, one row per thread --
  const int row = blockRow + tid;
  const float g0 = sgate[wave][lane][0];
  const float g1 = sgate[wave][lane][1];
  const float g2 = sgate[wave][lane][2];
  const float g3 = sgate[wave][lane][3];

  const float4 ef = qexpect(g0, g1, g2, g3, th_f);
  const float4 ei = qexpect(g0, g1, g2, g3, th_i);
  const float4 eg = qexpect(g0, g1, g2, g3, th_g);
  const float4 eo = qexpect(g0, g1, g2, g3, th_o);

  const float4 cv = *(const float4*)(cx + (size_t)row * QH);

  float4 cn, hn;
  cn.x = sig_(ef.x) * cv.x + sig_(ei.x) * tanh_(eg.x);
  cn.y = sig_(ef.y) * cv.y + sig_(ei.y) * tanh_(eg.y);
  cn.z = sig_(ef.z) * cv.z + sig_(ei.z) * tanh_(eg.z);
  cn.w = sig_(ef.w) * cv.w + sig_(ei.w) * tanh_(eg.w);
  hn.x = sig_(eo.x) * tanh_(cn.x);
  hn.y = sig_(eo.y) * tanh_(cn.y);
  hn.z = sig_(eo.z) * tanh_(cn.z);
  hn.w = sig_(eo.w) * tanh_(cn.w);

  *(float4*)(out + (size_t)row * QH)                       = hn;  // hx_new
  *(float4*)(out + (size_t)QB * QH + (size_t)row * QH)     = cn;  // cx_new
}

extern "C" void kernel_launch(void* const* d_in, const int* in_sizes, int n_in,
                              void* d_out, int out_size, void* d_ws, size_t ws_size,
                              hipStream_t stream) {
  (void)in_sizes; (void)n_in; (void)out_size; (void)d_ws; (void)ws_size;
  const float* x  = (const float*)d_in[0];
  const float* hx = (const float*)d_in[1];
  const float* cx = (const float*)d_in[2];
  const float* W  = (const float*)d_in[3];
  const float* b  = (const float*)d_in[4];
  const float* tf = (const float*)d_in[5];
  const float* ti = (const float*)d_in[6];
  const float* tg = (const float*)d_in[7];
  const float* to = (const float*)d_in[8];
  float* out = (float*)d_out;

  dim3 grid(QB / 256), block(256);
  hipLaunchKernelGGL(qlstm_kernel, grid, block, 0, stream,
                     x, hx, cx, W, b, tf, ti, tg, to, out);
}
